// GAT_76166950028493
// MI455X (gfx1250) — compile-verified, matrix operand order checked by reading, not run on hardware
//
#include <hip/hip_runtime.h>
#include <math.h>

typedef __attribute__((ext_vector_type(16))) _Float16 v16h;
typedef __attribute__((ext_vector_type(8)))  _Float16 v8h;
typedef __attribute__((ext_vector_type(8)))  float    v8f;

// ---------- fragment loader (CDNA5 16-bit A/B 16x32 layout) ----------
// lane L: l = L&15 (row/col), hi = L>>4.  Lane holds K = {k0+hi*8 .. +7} and
// {k0+16+hi*8 .. +7}; consecutive pairs pack into VGPRs -> v16h order.
// Each 8-half group is 16 bytes -> one b128 load.
__device__ __forceinline__ v16h frag_h(const _Float16* __restrict__ p) {
  v8h a = *(const v8h*)(p);
  v8h b = *(const v8h*)(p + 16);
  v16h r;
#pragma unroll
  for (int i = 0; i < 8; i++) { r[i] = a[i]; r[8 + i] = b[i]; }
  return r;
}

// ---------- WMMA GEMM: C[M x Dout] = act(A[M x K] @ W[Dout x K]^T + bias) ----
// A, W pre-converted to f16 (row-major).  Block = 8 waves; each wave owns a
// 16-row M tile; all waves share one 64-column W slab staged in LDS
// (row-padded by 8 halves -> conflict-free b128 LDS reads).
// CH: C stored f16.  act: 0 none, 1 ELU.
template<int CH>
__global__ void wmma_gemm(const _Float16* __restrict__ A, int lda,
                          const _Float16* __restrict__ W, int ldw,
                          const float* __restrict__ bias,
                          void* __restrict__ C_, int ldc,
                          int Mtiles, int K, int act) {
  __shared__ _Float16 wlds[64 * 264];   // 64 rows x (K<=256 + 8 pad) halves
  const int tid  = threadIdx.x;
  const int wave = tid >> 5;
  const int lane = tid & 31;
  const int tm   = blockIdx.x * (blockDim.x >> 5) + wave;
  const int l    = lane & 15;
  const int hi   = lane >> 4;
  const int n0   = blockIdx.y * 64;
  const int ks   = K + 8;               // padded LDS row stride (halves)

  // cooperative stage of the 64 x K weight slab into LDS (16B per thread/iter)
  const int chunks = K >> 3;
  for (int idx = tid; idx < 64 * chunks; idx += 256) {
    int r = idx / chunks, c = idx - r * chunks;
    *(v8h*)(&wlds[r * ks + c * 8]) = *(const v8h*)(W + (long)(n0 + r) * ldw + c * 8);
  }
  __syncthreads();

  if (tm >= Mtiles) return;
  const _Float16* pa = A + ((long)tm * 16 + l) * lda + hi * 8;

  v8f acc[4];
#pragma unroll
  for (int i = 0; i < 4; i++)
#pragma unroll
    for (int r = 0; r < 8; r++) acc[i][r] = 0.f;

  for (int k0 = 0; k0 < K; k0 += 32) {
    v16h af = frag_h(pa + k0);
#pragma unroll
    for (int t = 0; t < 4; t++) {
      // B fragment from LDS: row (t*16 + l), halves [k0+hi*8..+7], [k0+16+hi*8..+7]
      const _Float16* pb = &wlds[(t * 16 + l) * ks + k0 + hi * 8];
      v8h b0 = *(const v8h*)(pb);
      v8h b1 = *(const v8h*)(pb + 16);
      v16h bf;
#pragma unroll
      for (int i = 0; i < 8; i++) { bf[i] = b0[i]; bf[8 + i] = b1[i]; }
      acc[t] = __builtin_amdgcn_wmma_f32_16x16x32_f16(
          false, af, false, bf, (short)0, acc[t], false, false);
    }
  }
  // C/D layout: lane -> col n = l; VGPR r -> row hi*8 + r.
#pragma unroll
  for (int t = 0; t < 4; t++) {
    const int nn = n0 + t * 16 + l;
    const float bv = bias ? bias[nn] : 0.f;
#pragma unroll
    for (int r = 0; r < 8; r++) {
      const long mm = (long)tm * 16 + hi * 8 + r;
      float v = acc[t][r] + bv;
      if (act) v = v > 0.f ? v : __expf(v) - 1.f;
      if (CH) ((_Float16*)C_)[mm * ldc + nn] = (_Float16)v;
      else    ((float*)C_)[mm * ldc + nn] = v;
    }
  }
}

// ---------- misc small kernels ----------
__global__ void zero32(unsigned* __restrict__ p, long n) {
  long t = (long)blockIdx.x * blockDim.x + threadIdx.x;
  if (t < n) p[t] = 0u;
}

__global__ void f32to16(const float* __restrict__ in, _Float16* __restrict__ out, long n) {
  long t = (long)blockIdx.x * blockDim.x + threadIdx.x;
  if (t < n) out[t] = (_Float16)in[t];
}

// ec[e,j] = elu(PA[src[e],j] + PB[dst[e],j] + b[j]),  j < 64, f16 out
__global__ void edge_mlp0(const float* __restrict__ PA, const float* __restrict__ PB,
                          const float* __restrict__ b, const int* __restrict__ src,
                          const int* __restrict__ dst, _Float16* __restrict__ out, int E_) {
  long t = (long)blockIdx.x * blockDim.x + threadIdx.x;
  if (t >= (long)E_ * 64) return;
  int e = (int)(t >> 6);
  int j = (int)(t & 63);
  float v = PA[(long)src[e] * 64 + j] + PB[(long)dst[e] * 64 + j] + b[j];
  v = v > 0.f ? v : __expf(v) - 1.f;
  out[t] = (_Float16)v;
}

__global__ void node_scores(const float* __restrict__ ft, const float* __restrict__ al,
                            const float* __restrict__ ar, float* __restrict__ el,
                            float* __restrict__ er, int Nn, int H, int D) {
  long t = (long)blockIdx.x * blockDim.x + threadIdx.x;
  if (t >= (long)Nn * H) return;
  int n = (int)(t / H), h = (int)(t % H);
  const float* fp = ft + (long)n * H * D + (long)h * D;
  float sl = 0.f, sr = 0.f;
  for (int j = 0; j < D; j++) { float f = fp[j]; sl += f * al[h * D + j]; sr += f * ar[h * D + j]; }
  el[t] = sl; er[t] = sr;
}

__device__ __forceinline__ unsigned fkey(float f) {
  unsigned u = __float_as_uint(f);
  return (u & 0x80000000u) ? ~u : (u | 0x80000000u);
}
__device__ __forceinline__ float funkey(unsigned u) {
  unsigned v = (u & 0x80000000u) ? (u & 0x7fffffffu) : ~u;
  return __uint_as_float(v);
}

__global__ void edge_logits(const _Float16* __restrict__ ec, const _Float16* __restrict__ et,
                            const _Float16* __restrict__ ed, int F,
                            const float* __restrict__ ac, const float* __restrict__ at,
                            const float* __restrict__ ad,
                            const float* __restrict__ el, const float* __restrict__ er,
                            const int* __restrict__ src, const int* __restrict__ dst,
                            float* __restrict__ lg, unsigned* __restrict__ mkey,
                            int E_, int H) {
  long t = (long)blockIdx.x * blockDim.x + threadIdx.x;
  if (t >= (long)E_ * H) return;
  int e = (int)(t / H), h = (int)(t % H);
  int s = src[e], d = dst[e];
  float sum = el[(long)s * H + h] + er[(long)d * H + h];
  const _Float16* pc = ec + (long)e * F;
  const _Float16* pt = et + (long)e * F;
  const _Float16* pd = ed + (long)e * F;
  const float* wc = ac + (long)h * F;
  const float* wt = at + (long)h * F;
  const float* wd = ad + (long)h * F;
  for (int j = 0; j < F; j++)
    sum += (float)pc[j] * wc[j] + (float)pt[j] * wt[j] + (float)pd[j] * wd[j];
  float v = sum > 0.f ? sum : 0.2f * sum;   // leaky relu
  lg[t] = v;
  atomicMax(&mkey[(long)d * H + h], fkey(v));
}

__global__ void edge_exp(const float* __restrict__ lg, const unsigned* __restrict__ mkey,
                         const int* __restrict__ dst, float* __restrict__ exv,
                         float* __restrict__ den, int E_, int H) {
  long t = (long)blockIdx.x * blockDim.x + threadIdx.x;
  if (t >= (long)E_ * H) return;
  int e = (int)(t / H), h = (int)(t % H);
  int d = dst[e];
  float m = funkey(mkey[(long)d * H + h]);
  float v = __expf(lg[t] - m);
  exv[t] = v;
  atomicAdd(&den[(long)d * H + h], v);
}

__global__ void edge_message(const float* __restrict__ ft, const float* __restrict__ exv,
                             const float* __restrict__ den, const int* __restrict__ src,
                             const int* __restrict__ dst, float* __restrict__ rst,
                             int E_, int H, int D) {
  long t = (long)blockIdx.x * blockDim.x + threadIdx.x;
  int HD = H * D;
  if (t >= (long)E_ * HD) return;
  int j = (int)(t % HD);
  int e = (int)(t / HD);
  int h = j / D;
  int s = src[e], d = dst[e];
  float a = exv[(long)e * H + h] / den[(long)d * H + h];
  atomicAdd(&rst[(long)d * HD + j], ft[(long)s * HD + j] * a);
}

// h = elu(elu(rst)) stored as f16
__global__ void elu2h(const float* __restrict__ rst, _Float16* __restrict__ hn, long n) {
  long t = (long)blockIdx.x * blockDim.x + threadIdx.x;
  if (t >= n) return;
  float v = rst[t];
  v = v > 0.f ? v : __expf(v) - 1.f;
  v = v > 0.f ? v : __expf(v) - 1.f;
  hn[t] = (_Float16)v;
}

// out[n,f] = mean over 4 heads of rst[n, h*128 + f]
__global__ void mean_heads(const float* __restrict__ rst, float* __restrict__ out, int Nn) {
  long t = (long)blockIdx.x * blockDim.x + threadIdx.x;
  if (t >= (long)Nn * 128) return;
  int n = (int)(t >> 7), f = (int)(t & 127);
  const float* p = rst + (long)n * 512 + f;
  out[t] = 0.25f * (p[0] + p[128] + p[256] + p[384]);
}

// ---------- host ----------
extern "C" void kernel_launch(void* const* d_in, const int* in_sizes, int n_in,
                              void* d_out, int out_size, void* d_ws, size_t ws_size,
                              hipStream_t stream) {
  (void)in_sizes; (void)n_in; (void)out_size; (void)ws_size;
  const int N = 40000, E = 640000, H = 4;

  const float* x     = (const float*)d_in[0];
  const float* catf  = (const float*)d_in[1];
  const float* timef = (const float*)d_in[2];
  const float* sdf   = (const float*)d_in[3];
  const float* ddf   = (const float*)d_in[4];
  const int*   src   = (const int*)d_in[6];
  const int*   dst   = (const int*)d_in[7];

  const float *Wm[3][3], *bm[3][3];
  for (int m = 0; m < 3; m++)
    for (int s = 0; s < 3; s++) {
      Wm[m][s] = (const float*)d_in[8 + 6 * m + 2 * s];
      bm[m][s] = (const float*)d_in[8 + 6 * m + 2 * s + 1];
    }
  const float *gW[3], *gal[3], *gar[3], *gac[3], *gatt[3], *gad[3];
  for (int i = 0; i < 3; i++) {
    gW[i]   = (const float*)d_in[26 + 6 * i + 0];
    gal[i]  = (const float*)d_in[26 + 6 * i + 1];
    gar[i]  = (const float*)d_in[26 + 6 * i + 2];
    gac[i]  = (const float*)d_in[26 + 6 * i + 3];
    gatt[i] = (const float*)d_in[26 + 6 * i + 4];
    gad[i]  = (const float*)d_in[26 + 6 * i + 5];
  }

  char* wsp = (char*)d_ws;
  size_t off = 0;
  auto alloc = [&](size_t bytes) -> void* {
    void* p = wsp + off;
    off += (bytes + 255) & ~(size_t)255;
    return p;
  };

  // f16 copies of node features and all GEMM weights
  _Float16* nf16[5];
  for (int i = 0; i < 5; i++) nf16[i] = (_Float16*)alloc((size_t)N * 128 * 2);
  const long WmSz[3] = { 64L * 256, 64L * 64, 128L * 64 };
  _Float16* Wm16[3][3];
  for (int m = 0; m < 3; m++)
    for (int s = 0; s < 3; s++) Wm16[m][s] = (_Float16*)alloc((size_t)WmSz[s] * 2);
  const long gWSz[3] = { 256L * 128, 256L * 256, 512L * 256 };
  _Float16* gW16[3];
  for (int i = 0; i < 3; i++) gW16[i] = (_Float16*)alloc((size_t)gWSz[i] * 2);

  float* P[3][2];
  for (int m = 0; m < 3; m++) {
    P[m][0] = (float*)alloc((size_t)N * 64 * 4);
    P[m][1] = (float*)alloc((size_t)N * 64 * 4);
  }
  _Float16* bufA[3]; _Float16* bufB[3];
  for (int m = 0; m < 3; m++) {
    bufA[m] = (_Float16*)alloc((size_t)E * 64 * 2);
    bufB[m] = (_Float16*)alloc((size_t)E * 128 * 2);
  }
  float*    ft   = (float*)alloc((size_t)N * 512 * 4);
  float*    rst  = (float*)alloc((size_t)N * 512 * 4);
  _Float16* hbuf = (_Float16*)alloc((size_t)N * 256 * 2);
  float*    el   = (float*)alloc((size_t)N * H * 4);
  float*    er   = (float*)alloc((size_t)N * H * 4);
  unsigned* mkey = (unsigned*)alloc((size_t)N * H * 4);
  float*    den  = (float*)alloc((size_t)N * H * 4);
  float*    lg   = (float*)alloc((size_t)E * H * 4);
  float*    exv  = (float*)alloc((size_t)E * H * 4);

  auto blocks = [](long n) { return dim3((unsigned)((n + 255) / 256)); };
  auto zero = [&](void* p, long nwords) {
    zero32<<<blocks(nwords), 256, 0, stream>>>((unsigned*)p, nwords);
  };
  auto conv = [&](const float* in, _Float16* o, long n) {
    f32to16<<<blocks(n), 256, 0, stream>>>(in, o, n);
  };
  auto gemm = [&](const _Float16* A, int lda, const _Float16* W, int ldw,
                  const float* bias, void* C, int ldc, int ch,
                  int M, int K, int Dout, int act) {
    dim3 g((unsigned)((M / 16 + 7) / 8), (unsigned)(Dout / 64));
    dim3 b(256);
    if (ch) wmma_gemm<1><<<g, b, 0, stream>>>(A, lda, W, ldw, bias, C, ldc, M / 16, K, act);
    else    wmma_gemm<0><<<g, b, 0, stream>>>(A, lda, W, ldw, bias, C, ldc, M / 16, K, act);
  };

  // ---- pre-convert features and weights to f16 ----
  const float* nfSrc[5] = { x, catf, timef, sdf, ddf };
  for (int i = 0; i < 5; i++) conv(nfSrc[i], nf16[i], (long)N * 128);
  for (int m = 0; m < 3; m++)
    for (int s = 0; s < 3; s++) conv(Wm[m][s], Wm16[m][s], WmSz[s]);
  for (int i = 0; i < 3; i++) conv(gW[i], gW16[i], gWSz[i]);
  _Float16* x16   = nf16[0];
  _Float16* cf16  = nf16[1];
  _Float16* tf16  = nf16[2];
  _Float16* sdf16 = nf16[3];
  _Float16* ddf16 = nf16[4];

  // ---- layer-0 edge MLPs: concat-linear decomposed into node GEMMs ----
  const _Float16* sFeat[3] = { cf16, tf16, sdf16 };
  const _Float16* dFeat[3] = { cf16, tf16, ddf16 };
  for (int m = 0; m < 3; m++) {
    gemm(sFeat[m], 128, Wm16[m][0],       256, nullptr, P[m][0], 64, 0, N, 128, 64, 0);
    gemm(dFeat[m], 128, Wm16[m][0] + 128, 256, nullptr, P[m][1], 64, 0, N, 128, 64, 0);
  }
  for (int m = 0; m < 3; m++)
    edge_mlp0<<<blocks((long)E * 64), 256, 0, stream>>>(P[m][0], P[m][1], bm[m][0],
                                                        src, dst, bufB[m], E);

  // ---- one GAT layer (attention + aggregation; rst left un-activated) ----
  auto gat_layer = [&](const _Float16* hA, int K, int li,
                       const _Float16* ecb, const _Float16* etb, const _Float16* edb, int D) {
    const int HD = H * D;
    gemm(hA, K, gW16[li], K, nullptr, ft, HD, 0, N, K, HD, 0);
    node_scores<<<blocks((long)N * H), 256, 0, stream>>>(ft, gal[li], gar[li], el, er, N, H, D);
    zero(mkey, (long)N * H);
    zero(den, (long)N * H);
    edge_logits<<<blocks((long)E * H), 256, 0, stream>>>(ecb, etb, edb, D,
        gac[li], gatt[li], gad[li], el, er, src, dst, lg, mkey, E, H);
    edge_exp<<<blocks((long)E * H), 256, 0, stream>>>(lg, mkey, dst, exv, den, E, H);
    zero(rst, (long)N * HD);
    edge_message<<<blocks((long)E * HD), 256, 0, stream>>>(ft, exv, den, src, dst, rst, E, H, D);
  };

  // GAT layer 0 (h = x16, K=128)
  gat_layer(x16, 128, 0, bufB[0], bufB[1], bufB[2], 64);
  elu2h<<<blocks((long)N * 256), 256, 0, stream>>>(rst, hbuf, (long)N * 256);

  // edge MLP stage 1: B(64) -> A(64), linear+ELU
  for (int m = 0; m < 3; m++)
    gemm(bufB[m], 64, Wm16[m][1], 64, bm[m][1], bufA[m], 64, 1, E, 64, 64, 1);

  // GAT layer 1 (h f16, K=256)
  gat_layer(hbuf, 256, 1, bufA[0], bufA[1], bufA[2], 64);
  elu2h<<<blocks((long)N * 256), 256, 0, stream>>>(rst, hbuf, (long)N * 256);

  // edge MLP stage 2: A(64) -> B(128), linear+ELU
  for (int m = 0; m < 3; m++)
    gemm(bufA[m], 64, Wm16[m][2], 64, bm[m][2], bufB[m], 128, 1, E, 64, 128, 1);

  // final GAT layer (no activation), then mean over heads
  gat_layer(hbuf, 256, 2, bufB[0], bufB[1], bufB[2], 128);
  mean_heads<<<blocks((long)N * 128), 256, 0, stream>>>(rst, (float*)d_out, N);
}
